// AttnLayer_60464549593314
// MI455X (gfx1250) — matrix operand order repaired and checked by reading, not compiled, not run on hardware
//
#include <hip/hip_runtime.h>
#include <hip/hip_bf16.h>

// ---------------------------------------------------------------------------
// Problem constants (from reference): L_IN=1024, L_ENC=1024, B=16, D=512.
// ---------------------------------------------------------------------------
constexpr int LQ = 1024;      // L_IN
constexpr int LE = 1024;      // L_ENC
constexpr int NB = 16;        // batch
constexpr int DI = 512;       // D_IN
constexpr int DE = 512;       // D_ENC
constexpr int DM = 512;       // D_EMB
constexpr int ROWS = LQ * NB; // 16384 flattened (l,b) rows

typedef __bf16 bf16_t;
typedef __attribute__((ext_vector_type(16))) __bf16 v16bf;
typedef __attribute__((ext_vector_type(8)))  __bf16 v8bf;
typedef __attribute__((ext_vector_type(4)))  __bf16 v4bf;
typedef __attribute__((ext_vector_type(8)))  float  v8f;

// ---------------------------------------------------------------------------
// f32 -> bf16 convert (vectorized: global_load_b128 in, b64 out)
// ---------------------------------------------------------------------------
__global__ __launch_bounds__(256) void cvt_f32_bf16(const float* __restrict__ src,
                                                    bf16_t* __restrict__ dst,
                                                    long n4) {
  long i = (long)blockIdx.x * blockDim.x + threadIdx.x;
  long stride = (long)gridDim.x * blockDim.x;
  for (; i < n4; i += stride) {
    float4 v = ((const float4*)src)[i];
    v4bf o = { (__bf16)v.x, (__bf16)v.y, (__bf16)v.z, (__bf16)v.w };
    ((v4bf*)dst)[i] = o;
  }
}

// ---------------------------------------------------------------------------
// enc_seq (m,b,e) f32  ->  encT (b,e,m) bf16, LDS 32x32 tile transpose
// ---------------------------------------------------------------------------
__global__ __launch_bounds__(256) void enc_transpose(const float* __restrict__ enc,
                                                     bf16_t* __restrict__ encT) {
  __shared__ bf16_t tile[32][33];
  const int b = blockIdx.z;
#pragma unroll
  for (int j = 0; j < 4; ++j) {
    int ml = threadIdx.y + j * 8;
    int m  = blockIdx.x * 32 + ml;
    int e  = blockIdx.y * 32 + threadIdx.x;
    tile[ml][threadIdx.x] = (__bf16)enc[((long)m * NB + b) * DE + e];
  }
  __syncthreads();
#pragma unroll
  for (int j = 0; j < 4; ++j) {
    int el = threadIdx.y + j * 8;
    int e  = blockIdx.y * 32 + el;
    int m  = blockIdx.x * 32 + threadIdx.x;
    encT[((long)b * DE + e) * (long)LE + m] = tile[threadIdx.x][el];
  }
}

// ---------------------------------------------------------------------------
// Generic bf16 WMMA GEMM:  C[z] = A[z](MxK) * Bt[z](NxK)^T  (+ A2*B2t) (+bias)
//   - 256 threads = 8 wave32s; block tile 256(M) x 64(N); K-step 32
//   - B tile staged in LDS (stride 40 elems = conflict-free, b128-aligned),
//     shared by all 8 waves
//   - each wave: TWO A fragments (32 rows) x 4 B fragments -> 8 accumulators;
//     every ds_load'ed B fragment feeds two back-to-back v_wmma (one dscnt
//     wait per 2 matrix ops, 2x arithmetic intensity vs 1-A-frag version)
//   - A fragments loaded directly from global (contiguous b128 pairs)
// Fragment layouts per CDNA5 ISA 7.12.2 (16-bit A, B, f32 C/D).
// ---------------------------------------------------------------------------
template<bool DUAL, bool OUT_BF16, bool HAS_BIAS>
__global__ __launch_bounds__(256) void gemm_wmma_bf16(
    const bf16_t* __restrict__ A,  const bf16_t* __restrict__ A2,
    const bf16_t* __restrict__ Bt, const bf16_t* __restrict__ B2t,
    const float*  __restrict__ bias, const float* __restrict__ bias2,
    float* __restrict__ Cf, bf16_t* __restrict__ Cbf,
    int K, int lda, int ldb, int ldc,
    long strideAz, long strideBz, long strideCz) {
  constexpr int BK = 32;
  __shared__ bf16_t lb[2][64][40];   // 40-elem row stride: 80B, 16B-aligned

  const int tid  = threadIdx.x;
  const int wave = tid >> 5;
  const int lane = tid & 31;
  const int lrow = lane & 15;   // A-row / B-col / C-col within 16-tile
  const int half = lane >> 4;

  const long zA = (long)blockIdx.z * strideAz;
  const long zB = (long)blockIdx.z * strideBz;
  const long zC = (long)blockIdx.z * strideCz;

  const int M0 = blockIdx.x * 256 + wave * 32;  // this wave's 32 rows
  const int N0 = blockIdx.y * 64;               // block's 64 cols

  const v8f vzero = {0.f, 0.f, 0.f, 0.f, 0.f, 0.f, 0.f, 0.f};
  v8f acc[2][4];
#pragma unroll
  for (int s = 0; s < 2; ++s)
#pragma unroll
    for (int t = 0; t < 4; ++t) acc[s][t] = vzero;

  // staging assignment: thread -> (row n, 8-elem chunk kk) of the 64x32 B tile
  const int sn = tid >> 2;
  const int sk = (tid & 3) * 8;

  for (int k0 = 0; k0 < K; k0 += BK) {
    __syncthreads();
    {
      const bf16_t* bsrc = Bt + zB + (long)(N0 + sn) * ldb + k0 + sk;
      *(v8bf*)&lb[0][sn][sk] = *(const v8bf*)bsrc;
      if (k0 + BK < K) __builtin_prefetch(bsrc + BK, 0, 1);  // global_prefetch_b8
      if (DUAL) {
        const bf16_t* b2src = B2t + zB + (long)(N0 + sn) * ldb + k0 + sk;
        *(v8bf*)&lb[1][sn][sk] = *(const v8bf*)b2src;
      }
    }
    __syncthreads();

    // ---- two A fragments (16x32 bf16 each):
    //      elems 0..7 -> K = 8*half + i, elems 8..15 -> K = 16 + 8*half + i
    const bf16_t* arow0 = A + zA + (long)(M0 + lrow) * lda + k0;
    const bf16_t* arow1 = arow0 + (long)16 * lda;
    v16bf afrag0 = __builtin_shufflevector(
        *(const v8bf*)(arow0 + 8 * half), *(const v8bf*)(arow0 + 16 + 8 * half),
        0,1,2,3,4,5,6,7,8,9,10,11,12,13,14,15);
    v16bf afrag1 = __builtin_shufflevector(
        *(const v8bf*)(arow1 + 8 * half), *(const v8bf*)(arow1 + 16 + 8 * half),
        0,1,2,3,4,5,6,7,8,9,10,11,12,13,14,15);

#pragma unroll
    for (int t = 0; t < 4; ++t) {
      // ---- B fragment (32x16): col N=lrow, elems 0..7 -> K = 8*half + i, elems 8..15 -> K = 16 + 8*half + i
      const bf16_t* brow = &lb[0][t * 16 + lrow][8 * half];
      v16bf bfrag = __builtin_shufflevector(
          *(const v8bf*)brow, *(const v8bf*)(brow + 16),
          0,1,2,3,4,5,6,7,8,9,10,11,12,13,14,15);
      acc[0][t] = __builtin_amdgcn_wmma_f32_16x16x32_bf16(
          false, afrag0, false, bfrag, (short)0, acc[0][t], false, false);
      acc[1][t] = __builtin_amdgcn_wmma_f32_16x16x32_bf16(
          false, afrag1, false, bfrag, (short)0, acc[1][t], false, false);
    }

    if (DUAL) {
      const bf16_t* a2row0 = A2 + zA + (long)(M0 + lrow) * lda + k0;
      const bf16_t* a2row1 = a2row0 + (long)16 * lda;
      v16bf afrag2_0 = __builtin_shufflevector(
          *(const v8bf*)(a2row0 + 8 * half), *(const v8bf*)(a2row0 + 16 + 8 * half),
          0,1,2,3,4,5,6,7,8,9,10,11,12,13,14,15);
      v16bf afrag2_1 = __builtin_shufflevector(
          *(const v8bf*)(a2row1 + 8 * half), *(const v8bf*)(a2row1 + 16 + 8 * half),
          0,1,2,3,4,5,6,7,8,9,10,11,12,13,14,15);
#pragma unroll
      for (int t = 0; t < 4; ++t) {
        const bf16_t* brow = &lb[1][t * 16 + lrow][8 * half];
        v16bf bfrag = __builtin_shufflevector(
            *(const v8bf*)brow, *(const v8bf*)(brow + 16),
            0,1,2,3,4,5,6,7,8,9,10,11,12,13,14,15);
        acc[0][t] = __builtin_amdgcn_wmma_f32_16x16x32_bf16(
            false, afrag2_0, false, bfrag, (short)0, acc[0][t], false, false);
        acc[1][t] = __builtin_amdgcn_wmma_f32_16x16x32_bf16(
            false, afrag2_1, false, bfrag, (short)0, acc[1][t], false, false);
      }
    }
  }

  // ---- epilogue: C layout — lane col = lrow, VGPR r -> row 8*half + r
#pragma unroll
  for (int s = 0; s < 2; ++s) {
#pragma unroll
    for (int t = 0; t < 4; ++t) {
      const int col = N0 + t * 16 + lrow;
      float badd = 0.f;
      if (HAS_BIAS) {
        badd = bias[col];
        if (DUAL) badd += bias2[col];
      }
#pragma unroll
      for (int r = 0; r < 8; ++r) {
        const int row = M0 + s * 16 + half * 8 + r;
        const long off = zC + (long)row * ldc + col;
        float v = acc[s][t][r] + badd;
        if (OUT_BF16) Cbf[off] = (__bf16)v; else Cf[off] = v;
      }
    }
  }
}

// ---------------------------------------------------------------------------
// Sum-normalize over m (axis of length 1024), fused bf16 re-quantization:
// attn[b,l,m] = scores[b,l,m] / sum_m scores[b,l,m]
// ---------------------------------------------------------------------------
__global__ __launch_bounds__(256) void sum_normalize(const float* __restrict__ scores,
                                                     bf16_t* __restrict__ attn) {
  __shared__ float red[256];
  const long base = (long)blockIdx.x * LE;
  float s = 0.f;
  for (int i = threadIdx.x; i < LE; i += 256) s += scores[base + i];
  red[threadIdx.x] = s;
  __syncthreads();
  for (int k = 128; k > 0; k >>= 1) {
    if (threadIdx.x < k) red[threadIdx.x] += red[threadIdx.x + k];
    __syncthreads();
  }
  const float inv = 1.0f / red[0];
  for (int i = threadIdx.x; i < LE; i += 256)
    attn[base + i] = (__bf16)(scores[base + i] * inv);
}

// ---------------------------------------------------------------------------
// Host-side orchestration
// ---------------------------------------------------------------------------
static inline size_t align256(size_t x) { return (x + 255) & ~(size_t)255; }

extern "C" void kernel_launch(void* const* d_in, const int* in_sizes, int n_in,
                              void* d_out, int out_size, void* d_ws, size_t ws_size,
                              hipStream_t stream) {
  const float* in_seq   = (const float*)d_in[0];   // (L,B,DI)
  const float* enc_seq  = (const float*)d_in[1];   // (M,B,DE)
  const float* prev_tgt = (const float*)d_in[2];   // (L,B,DM)
  const float* W_in2enc = (const float*)d_in[3];   // (DE,DI)
  const float* b_in2enc = (const float*)d_in[4];
  const float* W_lab2enc= (const float*)d_in[5];   // (DE,DM)
  const float* b_lab2enc= (const float*)d_in[6];
  const float* W_enc2in = (const float*)d_in[7];   // (DI,DE)
  const float* b_enc2in = (const float*)d_in[8];
  float* out = (float*)d_out;                      // (L,B,DI)

  // ---- workspace carve-up (bf16 staging + f32 scores) ----
  char* ws = (char*)d_ws;
  size_t cur = 0;
  auto take = [&](size_t bytes) { size_t o = cur; cur = align256(cur + bytes); return o; };
  const size_t nLBD = (size_t)ROWS * DE;           // 8,388,608 elems
  bf16_t* wsIn   = (bf16_t*)(ws + take(nLBD * 2));             // in_seq bf16
  bf16_t* wsTgt  = (bf16_t*)(ws + take(nLBD * 2));             // prev_target bf16
  bf16_t* wsEnc  = (bf16_t*)(ws + take(nLBD * 2));             // enc (m,b,e) bf16
  bf16_t* wsEncT = (bf16_t*)(ws + take((size_t)NB * DE * LE * 2)); // enc^T (b,e,m)
  bf16_t* wsW1   = (bf16_t*)(ws + take((size_t)DE * DI * 2));
  bf16_t* wsW2   = (bf16_t*)(ws + take((size_t)DE * DM * 2));
  bf16_t* wsW3   = (bf16_t*)(ws + take((size_t)DI * DE * 2));
  bf16_t* wsDval = (bf16_t*)(ws + take(nLBD * 2));             // d_vals (l,b,e)
  float*  wsScor = (float*) (ws + take((size_t)NB * LQ * LE * 4)); // scores (b,l,m)
  bf16_t* wsAttn = (bf16_t*)(ws + take((size_t)NB * LQ * LE * 2)); // attn (b,l,m)
  bf16_t* wsCtx  = (bf16_t*)(ws + take(nLBD * 2));             // context (l,b,e)
  (void)ws_size; (void)in_sizes; (void)n_in; (void)out_size;

  // ---- 1) precision conversion (f32 -> bf16) ----
  {
    long n4 = (long)nLBD / 4;
    int grid = (int)((n4 + 255) / 256); if (grid > 4096) grid = 4096;
    cvt_f32_bf16<<<grid, 256, 0, stream>>>(in_seq,   wsIn,  n4);
    cvt_f32_bf16<<<grid, 256, 0, stream>>>(prev_tgt, wsTgt, n4);
    cvt_f32_bf16<<<grid, 256, 0, stream>>>(enc_seq,  wsEnc, n4);
    long w4 = (long)(DE * DI) / 4;
    int wgrid = (int)((w4 + 255) / 256);
    cvt_f32_bf16<<<wgrid, 256, 0, stream>>>(W_in2enc,  wsW1, w4);
    cvt_f32_bf16<<<wgrid, 256, 0, stream>>>(W_lab2enc, wsW2, w4);
    cvt_f32_bf16<<<wgrid, 256, 0, stream>>>(W_enc2in,  wsW3, w4);
  }
  // enc (m,b,e) -> encT (b,e,m)
  enc_transpose<<<dim3(LE / 32, DE / 32, NB), dim3(32, 8), 0, stream>>>(enc_seq, wsEncT);

  // ---- 2) d_vals = in_seq*W1^T + prev*W2^T + biases   (bf16 out, (l,b,e)) ----
  gemm_wmma_bf16<true, true, true><<<dim3(ROWS / 256, DE / 64, 1), 256, 0, stream>>>(
      wsIn, wsTgt, wsW1, wsW2, b_in2enc, b_lab2enc,
      nullptr, wsDval, /*K=*/DI, /*lda=*/DI, /*ldb=*/DI, /*ldc=*/DE, 0, 0, 0);

  // ---- 3) scores[b] = d_vals_b (LxE) * enc_b^T   (f32 out, (b,l,m)) ----
  gemm_wmma_bf16<false, false, false><<<dim3(LQ / 256, LE / 64, NB), 256, 0, stream>>>(
      wsDval, nullptr, wsEnc, nullptr, nullptr, nullptr,
      wsScor, nullptr, /*K=*/DE, /*lda=*/NB * DE, /*ldb=*/NB * DE, /*ldc=*/LE,
      /*strideAz=*/DE, /*strideBz=*/DE, /*strideCz=*/(long)LQ * LE);

  // ---- 4) attn = scores / sum_m scores  (fused bf16 requant) ----
  sum_normalize<<<NB * LQ, 256, 0, stream>>>(wsScor, wsAttn);

  // ---- 5) context[b] = attn_b (LxM) * encT_b^T  (bf16 out, (l,b,e)) ----
  gemm_wmma_bf16<false, true, false><<<dim3(LQ / 256, DE / 64, NB), 256, 0, stream>>>(
      wsAttn, nullptr, wsEncT, nullptr, nullptr, nullptr,
      nullptr, wsCtx, /*K=*/LE, /*lda=*/LE, /*ldb=*/LE, /*ldc=*/NB * DE,
      /*strideAz=*/(long)LQ * LE, /*strideBz=*/(long)DE * LE, /*strideCz=*/DE);

  // ---- 6) out = context * W_enc2in^T + bias  (f32 out, (l,b,i)) ----
  gemm_wmma_bf16<false, false, true><<<dim3(ROWS / 256, DI / 64, 1), 256, 0, stream>>>(
      wsCtx, nullptr, wsW3, nullptr, b_enc2in, nullptr,
      out, nullptr, /*K=*/DE, /*lda=*/DE, /*ldb=*/DE, /*ldc=*/DI, 0, 0, 0);
}
